// RBF_SVMLayer_71966472011991
// MI455X (gfx1250) — compile-verified, matrix operand order checked by reading, not run on hardware
//
#include <hip/hip_runtime.h>
#include <hip/hip_bf16.h>

// ---------------------------------------------------------------------------
// RBF-SVM decision function on MI455X (gfx1250, wave32, WMMA + TDM)
//   cross = X @ SV^T  via v_wmma_f32_16x16x32_bf16 (f32 accum)
//   dist  = |x|^2 + |s|^2 - 2*cross ;  k = exp(-dist/(2 sigma^2))
//   out   = sign(k @ alphas + bias)
// B tiles (pre-converted bf16 support vectors) are DMA'd global->LDS by the
// Tensor Data Mover, double-buffered, overlapped with the WMMA K-loop.
// ---------------------------------------------------------------------------

typedef __attribute__((ext_vector_type(4)))  __bf16 v4bf;
typedef __attribute__((ext_vector_type(8)))  __bf16 v8bf;
typedef __attribute__((ext_vector_type(16))) __bf16 v16bf;
typedef __attribute__((ext_vector_type(8)))  float  v8f;
typedef __attribute__((ext_vector_type(4)))  unsigned int u32x4;
typedef __attribute__((ext_vector_type(4)))  int  i32x4;
typedef __attribute__((ext_vector_type(8)))  int  i32x8;

#define DDIM 512
#define TN   64         // N rows per block (4 waves x 16 rows)
#define NWAVE 4
#define NTHREADS 128
#define TM   16         // M columns per chunk
#define LDSS 520        // bf16 elements per LDS row (1040 B: bank-conflict-free b128)

#if defined(__has_builtin)
#if __has_builtin(__builtin_amdgcn_tensor_load_to_lds)
#define HAVE_TDM 1
#endif
#endif

// ---- row squared-norm: one wave per row, 8 rows per 256-thread block -------
__global__ __launch_bounds__(256) void rownorm_kernel(const float* __restrict__ A,
                                                      float* __restrict__ out,
                                                      int rows) {
    const int lane = threadIdx.x & 31;
    const int wave = threadIdx.x >> 5;
    const int row  = blockIdx.x * 8 + wave;
    if (row >= rows) return;
    const float4* p = reinterpret_cast<const float4*>(A + (size_t)row * DDIM);
    float s = 0.f;
#pragma unroll
    for (int j = 0; j < 4; ++j) {
        float4 v = p[lane + 32 * j];
        s += v.x * v.x + v.y * v.y + v.z * v.z + v.w * v.w;
    }
#pragma unroll
    for (int off = 16; off >= 1; off >>= 1) s += __shfl_xor(s, off, 32);
    if (lane == 0) out[row] = s;
}

// ---- f32 -> bf16 bulk convert (support vectors, done once) -----------------
__global__ __launch_bounds__(256) void cvt_bf16_kernel(const float* __restrict__ in,
                                                       __bf16* __restrict__ out,
                                                       int n4) {     // n/4
    const int i = blockIdx.x * 256 + threadIdx.x;
    if (i >= n4) return;
    const float4 v = reinterpret_cast<const float4*>(in)[i];
    v4bf w = { (__bf16)v.x, (__bf16)v.y, (__bf16)v.z, (__bf16)v.w };
    *reinterpret_cast<v4bf*>(out + (size_t)i * 4) = w;
}

#ifdef HAVE_TDM
// Issue a TDM load of one 16x512 bf16 tile into LDS with a 16B pad every
// 1024B so LDS rows land at the 520-element stride the WMMA loads expect.
__device__ __forceinline__ void tdm_load_tile(const __bf16* gsrc, unsigned lds_off) {
    const unsigned long long ga = (unsigned long long)(uintptr_t)gsrc;
    u32x4 g0;
    g0[0] = 1u;                                   // count=1 (valid descriptor)
    g0[1] = lds_off;                              // lds_addr (bytes)
    g0[2] = (unsigned)(ga & 0xFFFFFFFFu);         // global_addr[31:0]
    g0[3] = (unsigned)((ga >> 32) & 0x1FFFFFFu)   // global_addr[56:32]
          | (2u << 30);                           // type=2 ("image")
    i32x8 g1;
    g1[0] = (int)((1u << 16)      // data_size = 1 -> 2-byte elements
                | (1u << 20)      // pad_enable
                | (7u << 22)      // pad_interval: 256 DWORDs (1024 B = one row)
                | (3u << 25));    // pad_amount: 4 DWORDs (16 B)
    g1[1] = (int)(512u << 16);    // tensor_dim0 = 512 (lo16)
    g1[2] = (int)(16u  << 16);    // tensor_dim0 hi = 0 | tensor_dim1 = 16 (lo16)
    g1[3] = (int)(512u << 16);    // tensor_dim1 hi = 0 | tile_dim0 = 512
    g1[4] = 16;                   // tile_dim1 = 16 | tile_dim2 = 0
    g1[5] = 512;                  // tensor_dim0_stride[31:0] = 512 elements
    g1[6] = 0;                    // stride hi / dim1_stride (unused, 2D)
    g1[7] = 0;
    i32x4 z4 = { 0, 0, 0, 0 };
#if defined(__clang_major__) && __clang_major__ >= 23
    i32x8 z8 = { 0, 0, 0, 0, 0, 0, 0, 0 };
    __builtin_amdgcn_tensor_load_to_lds(g0, g1, z4, z4, z8, 0);
#else
    __builtin_amdgcn_tensor_load_to_lds(g0, g1, z4, z4, 0);
#endif
}
#endif

// ---- main fused kernel -----------------------------------------------------
__global__ __launch_bounds__(NTHREADS, 1) void svm_wmma_kernel(
    const float* __restrict__ X, const __bf16* __restrict__ SVh,
    const float* __restrict__ alphas, const float* __restrict__ bias,
    const float* __restrict__ sigma, const float* __restrict__ xsq,
    const float* __restrict__ ssq, float* __restrict__ out, int M) {

    __shared__ __align__(16) __bf16 Abuf[TN * LDSS];      // 66,560 B
    __shared__ __align__(16) __bf16 Bbuf[2][TM * LDSS];   // 2 x 16,640 B

    const int tid   = threadIdx.x;
    const int lane  = tid & 31;
    const int wave  = tid >> 5;          // 0..3
    const int nbase = blockIdx.x * TN;

    // Stage A: convert this block's 64 X rows to bf16 in LDS (once).
    for (int i = tid; i < TN * (DDIM / 4); i += NTHREADS) {
        const int r  = i / (DDIM / 4);
        const int c4 = i % (DDIM / 4);
        const float4 v = reinterpret_cast<const float4*>(X + (size_t)(nbase + r) * DDIM)[c4];
        v4bf w = { (__bf16)v.x, (__bf16)v.y, (__bf16)v.z, (__bf16)v.w };
        *reinterpret_cast<v4bf*>(&Abuf[r * LDSS + c4 * 4]) = w;   // ds_store_b64
    }

    const int rsel = lane >> 4;          // 0|1 : which K-half this lane holds
    const int csel = lane & 15;          // WMMA row (A) / column (B) index
    const int wrow = wave * 16;          // wave's row base inside the tile

    float xsq8[8];
#pragma unroll
    for (int v = 0; v < 8; ++v) xsq8[v] = xsq[nbase + wrow + v + 8 * rsel];

    const float sg     = sigma[0];
    const float kscale = 1.4426950408889634f / (2.0f * sg * sg);  // log2(e)/(2s^2)

    float acc[8];
#pragma unroll
    for (int v = 0; v < 8; ++v) acc[v] = 0.f;

    const __bf16* arow = &Abuf[(wrow + csel) * LDSS];
    const int nchunks = M / TM;

#ifdef HAVE_TDM
    // Preload chunk 0 via the Tensor Data Mover.
    if (wave == 0) {
        tdm_load_tile(SVh, (unsigned)(uintptr_t)&Bbuf[0][0]);
        __builtin_amdgcn_s_wait_tensorcnt(0);
    }
    __syncthreads();   // A staged + chunk 0 resident
#else
    __syncthreads();   // A staged
#endif

    for (int ic = 0; ic < nchunks; ++ic) {
        const int mc  = ic * TM;
        const int cur = ic & 1;

#ifdef HAVE_TDM
        // Kick off DMA of the next chunk while we compute on this one.
        if (wave == 0 && ic + 1 < nchunks)
            tdm_load_tile(SVh + (size_t)(mc + TM) * DDIM,
                          (unsigned)(uintptr_t)&Bbuf[ic & 1 ? 0 : 1][0]);
#else
        // Fallback: cooperative LDS copy of the bf16 chunk (single buffer 0).
        for (int i = tid; i < TM * (DDIM / 8); i += NTHREADS) {
            const int r  = i / (DDIM / 8);
            const int c8 = i % (DDIM / 8);
            v8bf w = *reinterpret_cast<const v8bf*>(SVh + (size_t)(mc + r) * DDIM + c8 * 8);
            *reinterpret_cast<v8bf*>(&Bbuf[0][r * LDSS + c8 * 8]) = w;
        }
        __syncthreads();
#endif

        const __bf16* brow = &Bbuf[
#ifdef HAVE_TDM
            cur
#else
            0
#endif
        ][csel * LDSS];

        // 16x16 cross tile over K=512 in steps of 32.
        v8f c8 = {};
#pragma unroll 4
        for (int kb = 0; kb < DDIM; kb += 32) {
            const int o = kb + 8 * rsel;
            v8bf a0 = *reinterpret_cast<const v8bf*>(arow + o);        // ds_load_b128
            v8bf a1 = *reinterpret_cast<const v8bf*>(arow + o + 16);
            v8bf b0 = *reinterpret_cast<const v8bf*>(brow + o);
            v8bf b1 = *reinterpret_cast<const v8bf*>(brow + o + 16);
            v16bf a = __builtin_shufflevector(a0, a1, 0,1,2,3,4,5,6,7,8,9,10,11,12,13,14,15);
            v16bf b = __builtin_shufflevector(b0, b1, 0,1,2,3,4,5,6,7,8,9,10,11,12,13,14,15);
            c8 = __builtin_amdgcn_wmma_f32_16x16x32_bf16(false, a, false, b,
                                                         (short)0, c8, false, false);
        }

        // Epilogue: dist -> exp -> alpha-weighted accumulate (column = csel).
        const float al = alphas[mc + csel];
        const float sq = ssq[mc + csel];
#pragma unroll
        for (int v = 0; v < 8; ++v) {
            const float dist = xsq8[v] + sq - 2.0f * c8[v];
            acc[v] += al * __builtin_amdgcn_exp2f(-dist * kscale);
        }

#ifdef HAVE_TDM
        if (wave == 0) __builtin_amdgcn_s_wait_tensorcnt(0);  // next chunk landed
        __syncthreads();   // all waves done with cur buffer; next buffer ready
#else
        __syncthreads();
#endif
    }

    // Reduce the 16 columns held across lanes sharing a row group.
#pragma unroll
    for (int off = 8; off >= 1; off >>= 1)
#pragma unroll
        for (int v = 0; v < 8; ++v) acc[v] += __shfl_xor(acc[v], off, 32);

    if (csel == 0) {
        const float b0 = bias[0];
#pragma unroll
        for (int v = 0; v < 8; ++v) {
            const float d = acc[v] + b0;
            const float s = (d > 0.f) ? 1.f : ((d < 0.f) ? -1.f : d);
            out[nbase + wrow + v + 8 * rsel] = s;
        }
    }
}

// ---------------------------------------------------------------------------
extern "C" void kernel_launch(void* const* d_in, const int* in_sizes, int n_in,
                              void* d_out, int out_size, void* d_ws, size_t ws_size,
                              hipStream_t stream) {
    const float* X      = (const float*)d_in[0];
    const float* SV     = (const float*)d_in[1];
    const float* alphas = (const float*)d_in[2];
    const float* bias   = (const float*)d_in[3];
    const float* sigma  = (const float*)d_in[4];
    float* out = (float*)d_out;

    const int N = in_sizes[0] / DDIM;   // 16384
    const int M = in_sizes[1] / DDIM;   // 4096

    // Workspace layout: xsq[N] f32 | ssq[M] f32 | SVh[M*512] bf16
    float*  xsq = (float*)d_ws;
    float*  ssq = xsq + N;
    __bf16* SVh = (__bf16*)(ssq + M);

    rownorm_kernel<<<(N + 7) / 8, 256, 0, stream>>>(X,  xsq, N);
    rownorm_kernel<<<(M + 7) / 8, 256, 0, stream>>>(SV, ssq, M);
    const int n4 = (M * DDIM) / 4;
    cvt_bf16_kernel<<<(n4 + 255) / 256, 256, 0, stream>>>(SV, SVh, n4);

    svm_wmma_kernel<<<N / TN, NTHREADS, 0, stream>>>(X, SVh, alphas, bias, sigma,
                                                     xsq, ssq, out, M);
}
